// TransformerBlock_86887188398386
// MI455X (gfx1250) — matrix operand order repaired and checked by reading, not compiled, not run on hardware
//
#include <hip/hip_runtime.h>
#include <hip/hip_bf16.h>

// ---------------------------------------------------------------------------
// MI455X (gfx1250) transformer block. All GEMM-like math on v_wmma_f32_16x16x32_f16
// (f32 accumulate), fp32 for norms/rope/softmax/residuals.
// GEMM uses double-buffered GLOBAL_LOAD_ASYNC_TO_LDS_B128 (ASYNCcnt) prefetch.
// Workspace peak ~171 MB with buffer reuse.
// ---------------------------------------------------------------------------

typedef _Float16 f16;
typedef __attribute__((ext_vector_type(16))) _Float16 v16h;
typedef __attribute__((ext_vector_type(8)))  float    v8f;

union Frag16 { uint4 u[2]; v16h h; };

__device__ __forceinline__ v16h load_frag(const f16* p0, const f16* p1) {
  Frag16 f;
  f.u[0] = *(const uint4*)p0;
  f.u[1] = *(const uint4*)p1;
  return f.h;
}

__device__ __forceinline__ v8f wmma16(v16h a, v16h b, v8f c) {
  // D = A(16x32 f16) * B(32x16 f16) + C(16x16 f32)
  return __builtin_amdgcn_wmma_f32_16x16x32_f16(false, a, false, b, (short)0, c,
                                                false, false);
}

// Async memory->LDS DMA (CDNA5, tracked with ASYNCcnt). lds = group-segment
// byte offset (HW adds LDS_BASE), g = per-lane 64-bit global address, 16B/lane.
__device__ __forceinline__ void async_b128(unsigned int lds, const void* g) {
  asm volatile("global_load_async_to_lds_b128 %0, %1, off"
               :: "v"(lds), "v"(g) : "memory");
}
__device__ __forceinline__ void wait_async0() {
  asm volatile("s_wait_asynccnt 0x0" ::: "memory");
}

// ---------------------------------------------------------------------------
// Generic GEMM: C[M][N] = A[M][K](f16,row-major) * Bt[N][K](f16) (+ Res)
// Block tile 128(M) x 64(N), k-step 32, 8 waves: wave w owns rows w*16..w*16+15,
// all 4 n-tiles. LDS: 2 x (As 8KB + Bs 4KB) double-buffered, filled by
// global_load_async_to_lds_b128; one s_barrier per k-step.
// ---------------------------------------------------------------------------
__global__ __launch_bounds__(256) void gemm_nt(
    const f16* __restrict__ A, int lda,
    const f16* __restrict__ Bt, int ldb,
    float* __restrict__ C, int ldc,
    const float* __restrict__ Res, int K)
{
  __shared__ _Float16 As[2][128 * 32];
  __shared__ _Float16 Bs[2][64 * 32];
  const int tid  = threadIdx.x;
  const int wave = tid >> 5, lane = tid & 31;
  const int sub  = lane >> 4, l16 = lane & 15;     // half-wave, col/row-in-tile
  const int m0 = blockIdx.x * 128, n0 = blockIdx.y * 64;

  const unsigned int asBase = (unsigned int)(size_t)(void*)&As[0][0];
  const unsigned int bsBase = (unsigned int)(size_t)(void*)&Bs[0][0];

  // this thread's staging assignments (2 x 16B for As, 1 x 16B for Bs)
  const int ra0 = tid >> 2,         ca0 = (tid & 3) << 3;
  const int ra1 = (tid + 256) >> 2, ca1 = ((tid + 256) & 3) << 3;
  const int rb  = tid >> 2,         cb  = (tid & 3) << 3;

  auto issue = [&](int k0, int buf) {
    const unsigned int asb = asBase + (unsigned int)buf * (128 * 32 * 2);
    const unsigned int bsb = bsBase + (unsigned int)buf * (64 * 32 * 2);
    async_b128(asb + (unsigned int)(ra0 * 32 + ca0) * 2,
               &A[(size_t)(m0 + ra0) * lda + k0 + ca0]);
    async_b128(asb + (unsigned int)(ra1 * 32 + ca1) * 2,
               &A[(size_t)(m0 + ra1) * lda + k0 + ca1]);
    async_b128(bsb + (unsigned int)(rb * 32 + cb) * 2,
               &Bt[(size_t)(n0 + rb) * ldb + k0 + cb]);
  };

  v8f acc[4] = {};
  const int nsteps = K >> 5;

  issue(0, 0);
  for (int i = 0; i < nsteps; ++i) {
    const int cur = i & 1;
    wait_async0();          // own prefetch for buf[cur] complete
    __syncthreads();        // everyone's prefetch complete; prev compute done
    if (i + 1 < nsteps) issue((i + 1) << 5, cur ^ 1);

    // A frag: lane holds row m=wave*16+l16; k = sub*16+j  (contiguous)
    const f16* ap = &As[cur][(wave * 16 + l16) * 32];
    v16h a = load_frag(ap + sub * 16, ap + sub * 16 + 8);
#pragma unroll
    for (int nt = 0; nt < 4; ++nt) {
      // B frag: lane holds col n=nt*16+l16; k = sub*16+j  (contiguous)
      const f16* bp = &Bs[cur][(nt * 16 + l16) * 32 + sub * 16];
      v16h b = load_frag(bp, bp + 8);
      acc[nt] = wmma16(a, b, acc[nt]);
    }
  }

#pragma unroll
  for (int nt = 0; nt < 4; ++nt) {
    int col = n0 + nt * 16 + l16;
#pragma unroll
    for (int r = 0; r < 8; ++r) {                  // row = sub*8 + r (C layout)
      size_t off = (size_t)(m0 + wave * 16 + sub * 8 + r) * ldc + col;
      float v = acc[nt][r];
      if (Res) v += Res[off];
      C[off] = v;
    }
  }
}

// ---------------------------------------------------------------------------
// RMSNorm over D=1024, output f16.
// ---------------------------------------------------------------------------
__global__ __launch_bounds__(256) void rmsnorm_k(
    const float* __restrict__ X, const float* __restrict__ w, f16* __restrict__ Y)
{
  const int row = blockIdx.x;
  const float* x = X + (size_t)row * 1024;
  __shared__ float red[256];
  float s = 0.f;
  for (int i = threadIdx.x; i < 1024; i += 256) { float v = x[i]; s += v * v; }
  red[threadIdx.x] = s;
  __syncthreads();
  for (int st = 128; st > 0; st >>= 1) {
    if (threadIdx.x < st) red[threadIdx.x] += red[threadIdx.x + st];
    __syncthreads();
  }
  float inv = rsqrtf(red[0] * (1.f / 1024.f) + 1e-6f);
  f16* y = Y + (size_t)row * 1024;
  for (int i = threadIdx.x; i < 1024; i += 256) y[i] = (f16)(x[i] * inv * w[i]);
}

// ---------------------------------------------------------------------------
// RoPE segment: rotate pairs (i, i+half) with freq 10000^{-i/half}.
// ---------------------------------------------------------------------------
__device__ __forceinline__ void rope_seg(const float* src, f16* dst,
                                         int off, int dim, float pos, float scale)
{
  int half_ = dim >> 1;
  for (int i = 0; i < half_; ++i) {
    float fr  = __expf(-9.210340371976184f * (float)i / (float)half_);
    float ang = pos * fr;
    float c = cosf(ang), s = sinf(ang);
    float x1 = src[off + i], x2 = src[off + half_ + i];
    dst[off + i]         = (f16)((x1 * c - x2 * s) * scale);
    dst[off + half_ + i] = (f16)((x1 * s + x2 * c) * scale);
  }
}

// ---------------------------------------------------------------------------
// RoPE(3D) + scatter into per-head, tile-permuted token order.
// Q,K: [B*16][2048][64] f16 (Q pre-scaled by 1/8). Vt: [B*16][64][2048] f16.
// One thread per (b, token, head).
// ---------------------------------------------------------------------------
__global__ __launch_bounds__(256) void rope_scatter(
    const float* __restrict__ QKV, f16* __restrict__ Q,
    f16* __restrict__ Kd, f16* __restrict__ Vt)
{
  int g = blockIdx.x * 256 + threadIdx.x;          // [0, 65536)
  int b = g >> 15;
  int n = (g >> 4) & 2047;
  int head = g & 15;
  const float* src = QKV + (size_t)(b * 2048 + n) * 3072 + head * 64;
  int t = n >> 8, hh = (n >> 4) & 15, ww = n & 15;
  int tile = ((hh >> 3) << 1) | (ww >> 3);         // T//8==1 -> tile=(tH,tW)
  int pos  = ((t & 7) << 6) | ((hh & 7) << 3) | (ww & 7);
  int p = tile * 512 + pos;
  size_t bh = (size_t)(b * 16 + head);
  f16* qd = Q  + (bh * 2048 + p) * 64;
  f16* kd = Kd + (bh * 2048 + p) * 64;
  rope_seg(src, qd, 0, 16, (float)t, 0.125f);
  rope_seg(src, qd, 16, 24, (float)hh, 0.125f);
  rope_seg(src, qd, 40, 24, (float)ww, 0.125f);
  const float* ks = src + 1024;
  rope_seg(ks, kd, 0, 16, (float)t, 1.f);
  rope_seg(ks, kd, 16, 24, (float)hh, 1.f);
  rope_seg(ks, kd, 40, 24, (float)ww, 1.f);
  const float* vs = src + 2048;
  f16* vt = Vt + bh * 64 * 2048 + p;
  for (int d = 0; d < 64; ++d) vt[(size_t)d * 2048] = (f16)vs[d];
}

// ---------------------------------------------------------------------------
// Flash attention. Heads 0..7: kv span = the 512-token tile containing the
// query block. Heads 8..15: kv span = all 2048 tokens. Each wave handles 16
// queries; block = 8 waves = 128 queries. Grid = B*16 heads * 16 q-blocks.
// O: [B*2048][1024] f16, token order back in natural (t,h,w) layout.
// ---------------------------------------------------------------------------
__global__ __launch_bounds__(256) void attn_k(
    const f16* __restrict__ Q, const f16* __restrict__ Kd,
    const f16* __restrict__ Vt, f16* __restrict__ O)
{
  __shared__ _Float16 Ps[8][16 * 32];              // per-wave P staging (C->A relayout)
  const int bx = blockIdx.x;
  const int qb = bx & 15, h = (bx >> 4) & 15, b = bx >> 8;
  const int tid = threadIdx.x, wave = tid >> 5, lane = tid & 31;
  const int sub = lane >> 4, l16 = lane & 15;
  const size_t bh = (size_t)(b * 16 + h);
  const f16* qp = Q  + bh * 2048 * 64;
  const f16* kp = Kd + bh * 2048 * 64;
  const f16* vp = Vt + bh * 64 * 2048;
  const int p0 = qb * 128 + wave * 16;

  v16h qf0, qf1;                                   // Q 16x64 in two A-frags
  {
    const f16* r = qp + (size_t)(p0 + l16) * 64 + sub * 16;
    qf0 = load_frag(r, r + 8);
    qf1 = load_frag(r + 32, r + 40);
  }

  const int kv_lo = (h < 8) ? ((qb >> 2) * 512) : 0;
  const int kv_hi = (h < 8) ? (kv_lo + 512) : 2048;

  float mrow[8], lrow[8];
  v8f acc[4] = {};
#pragma unroll
  for (int r = 0; r < 8; ++r) { mrow[r] = -3.0e38f; lrow[r] = 0.f; }

  _Float16* ps = &Ps[wave][0];

  for (int kv = kv_lo; kv < kv_hi; kv += 32) {
    v8f s0 = {}, s1 = {};
    {
      const f16* kr0 = kp + (size_t)(kv + l16) * 64 + sub * 16;       // keys kv..kv+15
      const f16* kr1 = kp + (size_t)(kv + 16 + l16) * 64 + sub * 16;  // keys +16..+31
      v16h b00 = load_frag(kr0,      kr0 + 8);   // d 0..31
      v16h b01 = load_frag(kr0 + 32, kr0 + 40);  // d 32..63
      v16h b10 = load_frag(kr1,      kr1 + 8);
      v16h b11 = load_frag(kr1 + 32, kr1 + 40);
      s0 = wmma16(qf0, b00, s0);
      s0 = wmma16(qf1, b01, s0);
      s1 = wmma16(qf0, b10, s1);
      s1 = wmma16(qf1, b11, s1);
    }
    float fs[8];
#pragma unroll
    for (int r = 0; r < 8; ++r) {
      float mx = fmaxf(s0[r], s1[r]);
      mx = fmaxf(mx, __shfl_xor(mx, 1, 32));
      mx = fmaxf(mx, __shfl_xor(mx, 2, 32));
      mx = fmaxf(mx, __shfl_xor(mx, 4, 32));
      mx = fmaxf(mx, __shfl_xor(mx, 8, 32));
      float mnew = fmaxf(mrow[r], mx);
      float p0v = __expf(s0[r] - mnew);
      float p1v = __expf(s1[r] - mnew);
      float rs = p0v + p1v;
      rs += __shfl_xor(rs, 1, 32);
      rs += __shfl_xor(rs, 2, 32);
      rs += __shfl_xor(rs, 4, 32);
      rs += __shfl_xor(rs, 8, 32);
      float f = __expf(mrow[r] - mnew);
      lrow[r] = lrow[r] * f + rs;
      mrow[r] = mnew;
      fs[r] = f;
      int rr = sub * 8 + r;                        // C layout: row = sub*8+r, col = l16
      ps[rr * 32 + l16]      = (f16)p0v;
      ps[rr * 32 + 16 + l16] = (f16)p1v;
    }
#pragma unroll
    for (int dt = 0; dt < 4; ++dt)
#pragma unroll
      for (int r = 0; r < 8; ++r) acc[dt][r] *= fs[r];

    // P as A-frag (wave-private LDS; same-wave DS ops are ordered)
    const f16* pr = ps + l16 * 32 + sub * 16;
    v16h pf = load_frag(pr, pr + 8);
#pragma unroll
    for (int dt = 0; dt < 4; ++dt) {
      const f16* vr = vp + (size_t)(dt * 16 + l16) * 2048 + kv + sub * 16;
      v16h bf = load_frag(vr, vr + 8);
      acc[dt] = wmma16(pf, bf, acc[dt]);
    }
  }

  int nat[8];                                      // un-permute tokens
#pragma unroll
  for (int r = 0; r < 8; ++r) {
    int pq = p0 + sub * 8 + r;
    int tile = pq >> 9, pos = pq & 511;
    int t  = pos >> 6;
    int hh = (((tile >> 1) & 1) << 3) | ((pos >> 3) & 7);
    int ww = ((tile & 1) << 3) | (pos & 7);
    nat[r] = t * 256 + hh * 16 + ww;
  }
#pragma unroll
  for (int dt = 0; dt < 4; ++dt)
#pragma unroll
    for (int r = 0; r < 8; ++r) {
      float o = acc[dt][r] / lrow[r];
      O[(size_t)(b * 2048 + nat[r]) * 1024 + h * 64 + dt * 16 + l16] = (f16)o;
    }
}

// ---------------------------------------------------------------------------
// SwiGLU elementwise: A2 = silu(G) * U as f16; padded cols (>=2730) -> 0.
// ---------------------------------------------------------------------------
__global__ __launch_bounds__(256) void swiglu_k(
    const float* __restrict__ G, const float* __restrict__ U, f16* __restrict__ A2)
{
  size_t i = (size_t)blockIdx.x * 256 + threadIdx.x;
  int col = (int)(i % 2752);
  float g = G[i], u = U[i];
  float s = g / (1.f + __expf(-g));
  A2[i] = (col < 2730) ? (f16)(s * u) : (f16)0.f;
}

// ---------------------------------------------------------------------------
// Transpose + f32->f16 convert with zero padding:
// dst[r][c] = (c<srcRows && r<srcCols) ? src[c][r] : 0 ; dst is [dstRows][dstCols]
// ---------------------------------------------------------------------------
__global__ __launch_bounds__(256) void convt_k(
    const float* __restrict__ src, f16* __restrict__ dst,
    int dstRows, int dstCols, int srcRows, int srcCols)
{
  size_t i = (size_t)blockIdx.x * 256 + threadIdx.x;
  size_t total = (size_t)dstRows * dstCols;
  if (i >= total) return;
  int r = (int)(i / dstCols), c = (int)(i % dstCols);
  float v = (c < srcRows && r < srcCols) ? src[(size_t)c * srcCols + r] : 0.f;
  dst[i] = (f16)v;
}

// ---------------------------------------------------------------------------
extern "C" void kernel_launch(void* const* d_in, const int* in_sizes, int n_in,
                              void* d_out, int out_size, void* d_ws, size_t ws_size,
                              hipStream_t stream)
{
  (void)in_sizes; (void)n_in; (void)out_size; (void)ws_size;
  const float* x       = (const float*)d_in[0];
  const float* norm1_w = (const float*)d_in[1];
  const float* wqkv    = (const float*)d_in[2];
  const float* wo      = (const float*)d_in[3];
  const float* norm2_w = (const float*)d_in[4];
  const float* mlp_w   = (const float*)d_in[5];
  const float* mlp_v   = (const float*)d_in[6];
  const float* mlp_o   = (const float*)d_in[7];
  float* out = (float*)d_out;

  constexpr int M   = 4096;   // B * T*H*W tokens
  constexpr int Dm  = 1024;
  constexpr int D3  = 3072;
  constexpr int DH  = 2730;   // MLP hidden
  constexpr int DHP = 2752;   // padded to multiple of 32/64

  char* ws = (char*)d_ws;
  size_t off = 0;
  auto alloc = [&](size_t bytes) -> void* {
    void* p = ws + off;
    off = (off + bytes + 255) & ~(size_t)255;
    return p;
  };

  f16*   wqkvT = (f16*)alloc((size_t)D3 * Dm * 2);
  f16*   woT   = (f16*)alloc((size_t)Dm * Dm * 2);
  f16*   mlpwT = (f16*)alloc((size_t)DHP * Dm * 2);
  f16*   mlpvT = (f16*)alloc((size_t)DHP * Dm * 2);
  f16*   mlpoT = (f16*)alloc((size_t)Dm * DHP * 2);
  float* X2    = (float*)alloc((size_t)M * Dm * 4);       // attn residual output
  float* bufA  = (float*)alloc((size_t)M * D3 * 4);       // qkv f32 -> later G
  float* bufB  = (float*)alloc((size_t)M * DHP * 4);      // U
  f16*   bufC  = (f16*)alloc((size_t)3 * 2 * 16 * 2048 * 64 * 2); // q,k,vT -> later A2
  f16*   bufD  = (f16*)alloc((size_t)M * Dm * 2);         // hN -> O -> h2

  f16* qh = bufC;
  f16* kh = qh + (size_t)2 * 16 * 2048 * 64;
  f16* vT = kh + (size_t)2 * 16 * 2048 * 64;
  f16* A2 = bufC;
  float* G = bufA;
  float* U = bufB;

  // 1) weights: transpose + f16 convert (+ zero pad)
  convt_k<<<(D3 * Dm) / 256, 256, 0, stream>>>(wqkv, wqkvT, D3, Dm, Dm, D3);
  convt_k<<<(Dm * Dm) / 256, 256, 0, stream>>>(wo, woT, Dm, Dm, Dm, Dm);
  convt_k<<<(DHP * Dm) / 256, 256, 0, stream>>>(mlp_w, mlpwT, DHP, Dm, Dm, DH);
  convt_k<<<(DHP * Dm) / 256, 256, 0, stream>>>(mlp_v, mlpvT, DHP, Dm, Dm, DH);
  convt_k<<<(Dm * DHP) / 256, 256, 0, stream>>>(mlp_o, mlpoT, Dm, DHP, DH, Dm);

  // 2) hN = rmsnorm(x)
  rmsnorm_k<<<M, 256, 0, stream>>>(x, norm1_w, bufD);
  // 3) qkv = hN @ wqkv   (f32 out)
  gemm_nt<<<dim3(M / 128, D3 / 64), 256, 0, stream>>>(bufD, Dm, wqkvT, Dm,
                                                      bufA, D3, nullptr, Dm);
  // 4) rope + permute into q/k/vT (f16)
  rope_scatter<<<256, 256, 0, stream>>>(bufA, qh, kh, vT);
  // 5) attention -> O (reuse bufD)
  attn_k<<<2 * 16 * 16, 256, 0, stream>>>(qh, kh, vT, bufD);
  // 6) X2 = x + O @ wo
  gemm_nt<<<dim3(M / 128, Dm / 64), 256, 0, stream>>>(bufD, Dm, woT, Dm,
                                                      X2, Dm, x, Dm);
  // 7) h2 = rmsnorm(X2)
  rmsnorm_k<<<M, 256, 0, stream>>>(X2, norm2_w, bufD);
  // 8) G = h2 @ mlp_w ; U = h2 @ mlp_v   (padded N)
  gemm_nt<<<dim3(M / 128, DHP / 64), 256, 0, stream>>>(bufD, Dm, mlpwT, Dm,
                                                       G, DHP, nullptr, Dm);
  gemm_nt<<<dim3(M / 128, DHP / 64), 256, 0, stream>>>(bufD, Dm, mlpvT, Dm,
                                                       U, DHP, nullptr, Dm);
  // 9) A2 = silu(G) * U  (f16, padded cols zero)
  swiglu_k<<<(int)(((size_t)M * DHP) / 256), 256, 0, stream>>>(G, U, A2);
  // 10) out = X2 + A2 @ mlp_out
  gemm_nt<<<dim3(M / 128, Dm / 64), 256, 0, stream>>>(A2, DHP, mlpoT, DHP,
                                                      out, Dm, X2, DHP);
}
